// Gemma2DecoderLayer_5016521801818
// MI455X (gfx1250) — compile-verified, matrix operand order checked by reading, not run on hardware
//
#include <hip/hip_runtime.h>

// ---------------- problem constants ----------------
#define TT   4096
#define HH   2048
#define NHEADS 16
#define NKVH   8
#define HDIM   128
#define II   8192
#define WINN 1023
#define SCALE 0.0625f      // 256^-0.5
#define RMS_EPS 1e-6f
#define THETA 10000.0f

typedef __bf16 bf16;
typedef __bf16 bf16x16 __attribute__((ext_vector_type(16)));
typedef float  f32x8   __attribute__((ext_vector_type(8)));

// ---------------- fp32 -> bf16 transposed convert: dst[N][K] = src[K][N] ----------------
__global__ __launch_bounds__(256)
void f2bf_t_kernel(const float* __restrict__ src, bf16* __restrict__ dst, int K, int N) {
    __shared__ float tile[32][33];
    const int k0 = blockIdx.y * 32;
    const int n0 = blockIdx.x * 32;
    const int tx = threadIdx.x;   // 0..31
    const int ty = threadIdx.y;   // 0..7
#pragma unroll
    for (int i = 0; i < 4; ++i)
        tile[ty + i * 8][tx] = src[(size_t)(k0 + ty + i * 8) * N + n0 + tx];
    __syncthreads();
#pragma unroll
    for (int i = 0; i < 4; ++i)
        dst[(size_t)(n0 + ty + i * 8) * K + k0 + tx] = (bf16)tile[tx][ty + i * 8];
}

// ---------------- RMSNorm family ----------------
// mode 0: x=a+b; out_f=x (residual); out_b=rms(x,w) bf16
// mode 1: out_f = rms(a,w) + b            (fp32)
// mode 2: out_b = rms(a,w)                (bf16)
// mode 3: out_f = rms(a,w)                (fp32)
__global__ __launch_bounds__(256)
void rms_kernel(const float* __restrict__ a, const float* __restrict__ b,
                const float* __restrict__ w, float* __restrict__ out_f,
                bf16* __restrict__ out_b, int mode) {
    __shared__ float red[256];
    const int row = blockIdx.x;
    const int t = threadIdx.x;
    float x[8];
    float ss = 0.f;
#pragma unroll
    for (int i = 0; i < 8; ++i) {
        const int c = t + i * 256;
        float v = a[(size_t)row * HH + c];
        if (mode == 0) v += b[(size_t)row * HH + c];
        x[i] = v;
        ss += v * v;
    }
    if (mode == 0) {
#pragma unroll
        for (int i = 0; i < 8; ++i)
            out_f[(size_t)row * HH + t + i * 256] = x[i];
    }
    red[t] = ss;
    __syncthreads();
    for (int s = 128; s > 0; s >>= 1) {
        if (t < s) red[t] += red[t + s];
        __syncthreads();
    }
    const float inv = rsqrtf(red[0] * (1.0f / HH) + RMS_EPS);
#pragma unroll
    for (int i = 0; i < 8; ++i) {
        const int c = t + i * 256;
        float y = x[i] * inv * (1.0f + w[c]);
        if (mode == 1) y += b[(size_t)row * HH + c];
        if (mode == 0 || mode == 2) out_b[(size_t)row * HH + c] = (bf16)y;
        if (mode == 1 || mode == 3) out_f[(size_t)row * HH + c] = y;
    }
}

// ---- WMMA bf16 GEMM: C[M,N] = A[M,K] * Bt[N,K]^T (+bias), double-buffered LDS ----
#define BM 128
#define BN 128
#define BK 32
#define LDT 40    // padded LDS stride (elements) for both tiles

__global__ __launch_bounds__(256)
void gemm_bf16_wmma(const bf16* __restrict__ A, const bf16* __restrict__ Bt,
                    float* __restrict__ C, const float* __restrict__ bias,
                    int M, int N, int K) {
    __shared__ __align__(16) bf16 As[2][BM * LDT];
    __shared__ __align__(16) bf16 Bs[2][BN * LDT];

    const int tid    = threadIdx.x;
    const int wave   = tid >> 5;
    const int lane   = tid & 31;
    const int lane16 = lane & 15;
    const int hlf    = lane >> 4;

    const int blockRow = blockIdx.y * BM;
    const int blockCol = blockIdx.x * BN;
    const int waveRow  = (wave >> 2) * 64;   // 0 / 64
    const int waveCol  = (wave & 3) * 32;    // 0,32,64,96

    f32x8 acc[4][2];
#pragma unroll
    for (int a = 0; a < 4; ++a)
#pragma unroll
        for (int b = 0; b < 2; ++b)
#pragma unroll
            for (int e = 0; e < 8; ++e) acc[a][b][e] = 0.0f;

    // per-thread tile-load coords: each thread loads 16 contiguous K-elems of one row
    const int trow = tid >> 1;         // 0..127 (A row within tile / Bt col within tile)
    const int tcol = (tid & 1) * 16;   // 0/16 within BK

    const bf16* aSrc = A  + (size_t)(blockRow + trow) * K + tcol;
    const bf16* bSrc = Bt + (size_t)(blockCol + trow) * K + tcol;

    // preload tile 0
    {
        uint4 a0 = *(const uint4*)aSrc;
        uint4 a1 = *(const uint4*)(aSrc + 8);
        uint4 b0 = *(const uint4*)bSrc;
        uint4 b1 = *(const uint4*)(bSrc + 8);
        *(uint4*)&As[0][trow * LDT + tcol]     = a0;
        *(uint4*)&As[0][trow * LDT + tcol + 8] = a1;
        *(uint4*)&Bs[0][trow * LDT + tcol]     = b0;
        *(uint4*)&Bs[0][trow * LDT + tcol + 8] = b1;
    }
    __syncthreads();

    int cur = 0;
    for (int k0 = 0; k0 < K; k0 += BK) {
        const bool has_next = (k0 + BK) < K;
        uint4 a0, a1, b0, b1;
        if (has_next) {
            a0 = *(const uint4*)(aSrc + k0 + BK);
            a1 = *(const uint4*)(aSrc + k0 + BK + 8);
            b0 = *(const uint4*)(bSrc + k0 + BK);
            b1 = *(const uint4*)(bSrc + k0 + BK + 8);
            if (k0 + 2 * BK < K) {     // warm L2 two tiles ahead (global_prefetch_b8)
                __builtin_prefetch(aSrc + k0 + 2 * BK, 0, 3);
                __builtin_prefetch(bSrc + k0 + 2 * BK, 0, 3);
            }
        }

        union Frag { unsigned int u[8]; bf16x16 v; };
        Frag bfrag[2];
#pragma unroll
        for (int tc = 0; tc < 2; ++tc) {
            const int col = waveCol + tc * 16 + lane16;
            *(uint4*)&bfrag[tc].u[0] = *(const uint4*)&Bs[cur][col * LDT + hlf * 8];
            *(uint4*)&bfrag[tc].u[4] = *(const uint4*)&Bs[cur][col * LDT + hlf * 8 + 16];
        }
#pragma unroll
        for (int tr = 0; tr < 4; ++tr) {
            Frag afrag;
            const int row = waveRow + tr * 16 + lane16;
            *(uint4*)&afrag.u[0] = *(const uint4*)&As[cur][row * LDT + hlf * 8];
            *(uint4*)&afrag.u[4] = *(const uint4*)&As[cur][row * LDT + hlf * 8 + 16];
#pragma unroll
            for (int tc = 0; tc < 2; ++tc) {
                acc[tr][tc] = __builtin_amdgcn_wmma_f32_16x16x32_bf16(
                    false, afrag.v, false, bfrag[tc].v,
                    (short)0, acc[tr][tc], false, false);
            }
        }

        if (has_next) {
            const int nxt = cur ^ 1;
            *(uint4*)&As[nxt][trow * LDT + tcol]     = a0;
            *(uint4*)&As[nxt][trow * LDT + tcol + 8] = a1;
            *(uint4*)&Bs[nxt][trow * LDT + tcol]     = b0;
            *(uint4*)&Bs[nxt][trow * LDT + tcol + 8] = b1;
        }
        __syncthreads();
        cur ^= 1;
    }

    // C/D layout: VGPR e, lanes 0-15 -> M=e, lanes 16-31 -> M=e+8; N = lane%16
#pragma unroll
    for (int tr = 0; tr < 4; ++tr) {
#pragma unroll
        for (int tc = 0; tc < 2; ++tc) {
            const int col = blockCol + waveCol + tc * 16 + lane16;
            const float bv = bias ? bias[col] : 0.0f;
#pragma unroll
            for (int e = 0; e < 8; ++e) {
                const int row = blockRow + waveRow + tr * 16 + e + hlf * 8;
                C[(size_t)row * N + col] = acc[tr][tc][e] + bv;
            }
        }
    }
}

// ---------------- RoPE (in-place, fp32) ----------------
__global__ __launch_bounds__(256)
void rope_kernel(float* __restrict__ x, const int* __restrict__ positions, int nheads) {
    int idx = blockIdx.x * blockDim.x + threadIdx.x;
    int d  = idx & 63;
    int hh = (idx >> 6) % nheads;
    int t  = idx / (64 * nheads);
    if (t >= TT) return;
    float pos = (float)positions[t];
    float inv_freq = powf(THETA, -(2.0f * d) / (float)HDIM);
    float f = pos * inv_freq;
    float c = cosf(f), s = sinf(f);
    size_t base = ((size_t)t * nheads + hh) * HDIM;
    float x1 = x[base + d];
    float x2 = x[base + d + 64];
    x[base + d]      = x1 * c - x2 * s;
    x[base + d + 64] = x2 * c + x1 * s;
}

// ---------------- sliding-window attention (online softmax) ----------------
__global__ __launch_bounds__(128)
void attn_kernel(const float* __restrict__ q, const float* __restrict__ k,
                 const float* __restrict__ v, const int* __restrict__ positions,
                 bf16* __restrict__ out) {
    __shared__ float qsh[HDIM];
    __shared__ float sc[128];
    __shared__ float red[128];
    const int i = blockIdx.x;
    const int h = blockIdx.y;
    const int kvh = h / (NHEADS / NKVH);
    const int t = threadIdx.x;

    qsh[t] = q[((size_t)i * NHEADS + h) * HDIM + t];
    __syncthreads();

    const int pi = positions[i];
    int jstart = pi - WINN; if (jstart < 0) jstart = 0;
    const int jend = pi;

    float m = -1e30f, l = 0.0f, acc = 0.0f;
    for (int j0 = jstart; j0 <= jend; j0 += 128) {
        int jn = jend - j0 + 1; if (jn > 128) jn = 128;
        float s = -1e30f;
        if (t < jn) {
            const float* kp = k + ((size_t)(j0 + t) * NKVH + kvh) * HDIM;
            float dot = 0.0f;
#pragma unroll 8
            for (int d = 0; d < HDIM; ++d) dot += qsh[d] * kp[d];
            s = dot * SCALE;
        }
        red[t] = s;
        __syncthreads();
        for (int r = 64; r > 0; r >>= 1) {
            if (t < r) red[t] = fmaxf(red[t], red[t + r]);
            __syncthreads();
        }
        const float cm = red[0];
        __syncthreads();
        const float nm = fmaxf(m, cm);
        const float p = (t < jn) ? expf(s - nm) : 0.0f;
        sc[t]  = p;
        red[t] = p;
        __syncthreads();
        for (int r = 64; r > 0; r >>= 1) {
            if (t < r) red[t] += red[t + r];
            __syncthreads();
        }
        const float psum = red[0];
        __syncthreads();
        const float scale = expf(m - nm);
        float a2 = acc * scale;
        for (int j = 0; j < jn; ++j)
            a2 += sc[j] * v[((size_t)(j0 + j) * NKVH + kvh) * HDIM + t];
        acc = a2;
        l = l * scale + psum;
        m = nm;
        __syncthreads();
    }
    out[((size_t)i * NHEADS + h) * HDIM + t] = (bf16)(acc / l);
}

// ---------------- up * gelu(gate) -> bf16 ----------------
__global__ __launch_bounds__(256)
void glu_kernel(const float* __restrict__ up, const float* __restrict__ gate,
                bf16* __restrict__ out, size_t n) {
    size_t i = (size_t)blockIdx.x * blockDim.x + threadIdx.x;
    if (i >= n) return;
    float g = gate[i];
    float gl = 0.5f * g * (1.0f + tanhf(0.7978845608f * (g + 0.044715f * g * g * g)));
    out[i] = (bf16)(up[i] * gl);
}

// ---------------- launcher ----------------
extern "C" void kernel_launch(void* const* d_in, const int* in_sizes, int n_in,
                              void* d_out, int out_size, void* d_ws, size_t ws_size,
                              hipStream_t stream) {
    const float* hs    = (const float*)d_in[0];
    const float* resi  = (const float*)d_in[1];
    const float* wq    = (const float*)d_in[2];
    const float* bq    = (const float*)d_in[3];
    const float* wk    = (const float*)d_in[4];
    const float* bk    = (const float*)d_in[5];
    const float* wv    = (const float*)d_in[6];
    const float* bv    = (const float*)d_in[7];
    const float* wo    = (const float*)d_in[8];
    const float* wg    = (const float*)d_in[9];
    const float* wu    = (const float*)d_in[10];
    const float* wd    = (const float*)d_in[11];
    const float* ln_in = (const float*)d_in[12];
    const float* ln_pa = (const float*)d_in[13];
    const float* ln_pf = (const float*)d_in[14];
    const float* ln_po = (const float*)d_in[15];
    const int*   pos   = (const int*)d_in[16];

    float* out_f32  = (float*)d_out;                     // out (T,H)
    float* res2_f32 = (float*)d_out + (size_t)TT * HH;   // res2 (T,H)

    char* ws = (char*)d_ws;
    size_t off = 0;
    auto alloc = [&](size_t bytes) -> void* {
        void* p = ws + off;
        off += (bytes + 255) & ~(size_t)255;
        return p;
    };

    const size_t nQW = (size_t)HH * (NHEADS * HDIM);   // 2048*2048
    const size_t nKW = (size_t)HH * (NKVH * HDIM);     // 2048*1024
    const size_t nMW = (size_t)HH * II;                // 2048*8192

    bf16* wq_b = (bf16*)alloc(nQW * 2);   // [N=2048][K=2048]
    bf16* wk_b = (bf16*)alloc(nKW * 2);   // [N=1024][K=2048]
    bf16* wv_b = (bf16*)alloc(nKW * 2);   // [N=1024][K=2048]
    bf16* wo_b = (bf16*)alloc(nQW * 2);   // [N=2048][K=2048]
    bf16* wg_b = (bf16*)alloc(nMW * 2);   // [N=8192][K=2048]
    bf16* wu_b = (bf16*)alloc(nMW * 2);   // [N=8192][K=2048]
    bf16* wd_b = (bf16*)alloc(nMW * 2);   // [N=2048][K=8192]

    float* res_f   = (float*)alloc((size_t)TT * HH * 4);
    bf16*  h_b     = (bf16*)alloc((size_t)TT * HH * 2);
    float* q_f     = (float*)alloc((size_t)TT * NHEADS * HDIM * 4);
    float* k_f     = (float*)alloc((size_t)TT * NKVH * HDIM * 4);
    float* v_f     = (float*)alloc((size_t)TT * NKVH * HDIM * 4);
    bf16*  attn_b  = (bf16*)alloc((size_t)TT * HH * 2);
    float* attno_f = (float*)alloc((size_t)TT * HH * 4);
    bf16*  h2_b    = (bf16*)alloc((size_t)TT * HH * 2);
    float* gate_f  = (float*)alloc((size_t)TT * II * 4);
    float* up_f    = (float*)alloc((size_t)TT * II * 4);
    bf16*  act_b   = (bf16*)alloc((size_t)TT * II * 2);
    float* mlp_f   = (float*)alloc((size_t)TT * HH * 4);

    // transposing converts: src [K][N] fp32 -> dst [N][K] bf16
    auto cvt_t = [&](const float* s, bf16* d, int K, int N) {
        f2bf_t_kernel<<<dim3(N / 32, K / 32), dim3(32, 8), 0, stream>>>(s, d, K, N);
    };
    auto gemm = [&](const bf16* A, const bf16* Bt, float* C, const float* bias,
                    int M, int N, int K) {
        gemm_bf16_wmma<<<dim3(N / BN, M / BM), dim3(256), 0, stream>>>(A, Bt, C, bias, M, N, K);
    };

    // 1) convert+transpose weights to bf16
    cvt_t(wq, wq_b, HH, NHEADS * HDIM);
    cvt_t(wk, wk_b, HH, NKVH * HDIM);
    cvt_t(wv, wv_b, HH, NKVH * HDIM);
    cvt_t(wo, wo_b, HH, HH);
    cvt_t(wg, wg_b, HH, II);
    cvt_t(wu, wu_b, HH, II);
    cvt_t(wd, wd_b, II, HH);

    // 2) x = hs + residual; res = x; h = rms(x, ln_in) -> bf16
    rms_kernel<<<dim3(TT), dim3(256), 0, stream>>>(hs, resi, ln_in, res_f, h_b, 0);

    // 3) QKV projections
    gemm(h_b, wq_b, q_f, bq, TT, NHEADS * HDIM, HH);
    gemm(h_b, wk_b, k_f, bk, TT, NKVH * HDIM, HH);
    gemm(h_b, wv_b, v_f, bv, TT, NKVH * HDIM, HH);

    // 4) RoPE on q and k (in place)
    rope_kernel<<<dim3((TT * NHEADS * 64) / 256), dim3(256), 0, stream>>>(q_f, pos, NHEADS);
    rope_kernel<<<dim3((TT * NKVH * 64) / 256), dim3(256), 0, stream>>>(k_f, pos, NKVH);

    // 5) sliding-window attention -> bf16 activations
    attn_kernel<<<dim3(TT, NHEADS), dim3(128), 0, stream>>>(q_f, k_f, v_f, pos, attn_b);

    // 6) output projection
    gemm(attn_b, wo_b, attno_f, nullptr, TT, HH, HH);

    // 7) res2 = rms(attn_out, ln_post_attn) + res ; h2 = rms(res2, ln_pre_ff)
    rms_kernel<<<dim3(TT), dim3(256), 0, stream>>>(attno_f, res_f, ln_pa, res2_f32, nullptr, 1);
    rms_kernel<<<dim3(TT), dim3(256), 0, stream>>>(res2_f32, nullptr, ln_pf, nullptr, h2_b, 2);

    // 8) MLP
    gemm(h2_b, wg_b, gate_f, nullptr, TT, II, HH);
    gemm(h2_b, wu_b, up_f,   nullptr, TT, II, HH);
    glu_kernel<<<dim3((unsigned)(((size_t)TT * II + 255) / 256)), dim3(256), 0, stream>>>(
        up_f, gate_f, act_b, (size_t)TT * II);
    gemm(act_b, wd_b, mlp_f, nullptr, TT, HH, II);

    // 9) out = rms(mlp, ln_post_ff)
    rms_kernel<<<dim3(TT), dim3(256), 0, stream>>>(mlp_f, nullptr, ln_po, out_f32, nullptr, 3);
}
